// FuCont_ASPP_PC_11733850652680
// MI455X (gfx1250) — matrix operand order repaired and checked by reading, not compiled, not run on hardware
//
#include <hip/hip_runtime.h>

// ---------------------------------------------------------------------------
// CDNA5 (gfx1250): bf16 WMMA for all GEMM-shaped work (implicit-GEMM convs,
// CAM energy/out GEMMs) with double-buffered LDS stages and
// GLOBAL_LOAD_ASYNC_TO_LDS_B128 tile feeds (ASYNCcnt path).
// fp32 VALU for the bandwidth-bound reductions (GN, L2 norms, softmax, CC).
// Problem constants: B=4, C=256, Cq=64, H=W=128, HW=16384, NPIX=65536.
// ---------------------------------------------------------------------------

typedef __attribute__((ext_vector_type(16))) __bf16 v16bf;
typedef __attribute__((ext_vector_type(8)))  __bf16 v8bf;
typedef __attribute__((ext_vector_type(8)))  float  v8f;
typedef __attribute__((ext_vector_type(4)))  int    v4i;

#define LDT 40  // padded LDS row stride (bf16 elems) for 64/128 x 32 tiles

// ---- async LDS tile feed (gfx1250 ASYNCcnt path), guarded -----------------
#if defined(__AMDGCN__) && __has_builtin(__builtin_amdgcn_global_load_async_to_lds_b128)
#define HAVE_ASYNC_LDS 1
typedef __attribute__((address_space(1))) v4i v4i_g;   // global AS int4
typedef __attribute__((address_space(3))) v4i v4i_l;   // LDS AS int4
__device__ __forceinline__ void async_copy16(const __bf16* g, __bf16* l) {
  __builtin_amdgcn_global_load_async_to_lds_b128((v4i_g*)g, (v4i_l*)l, 0, 0);
}
__device__ __forceinline__ void async_wait0() {
#if __has_builtin(__builtin_amdgcn_s_wait_asynccnt)
  __builtin_amdgcn_s_wait_asynccnt(0);
#else
  asm volatile("s_wait_asynccnt 0x0" ::: "memory");
#endif
}
#else
#define HAVE_ASYNC_LDS 0
__device__ __forceinline__ void async_copy16(const __bf16* g, __bf16* l) {
  *reinterpret_cast<v8bf*>(l) = *reinterpret_cast<const v8bf*>(g);
}
__device__ __forceinline__ void async_wait0() {}
#endif

// ---- WMMA fragment helpers (ISA 7.12.2 layouts, wave32) -------------------
// A 16x32 bf16: lane r=lane&15 holds row M=r; K = (e<8?0:16) + (lane>>4)*8 + (e&7)
__device__ __forceinline__ v16bf frag_a(const __bf16* rowp, int hi) {
  v8bf lo = *reinterpret_cast<const v8bf*>(rowp + hi * 8);
  v8bf hv = *reinterpret_cast<const v8bf*>(rowp + 16 + hi * 8);
  v16bf a;
#pragma unroll
  for (int i = 0; i < 8; ++i) { a[i] = lo[i]; a[i + 8] = hv[i]; }
  return a;
}
// B 32x16 bf16 (stored transposed [n][k] in LDS): lane r holds col N=r;
// K = (e<8?0:16) + (lane>>4)*8 + (e&7)  (same order as A)
__device__ __forceinline__ v16bf frag_b(const __bf16* rowp, int hi) {
  v8bf b0 = *reinterpret_cast<const v8bf*>(rowp + hi * 8);
  v8bf b1 = *reinterpret_cast<const v8bf*>(rowp + 16 + hi * 8);
  v16bf b;
#pragma unroll
  for (int i = 0; i < 8; ++i) { b[i] = b0[i]; b[i + 8] = b1[i]; }
  return b;
}

// One K=32 stage: 8 waves; wave (wm = wave&3, wn = wave>>2) owns a 16x64
// strip -> 1 A fragment feeds 4 WMMAs (4 f32 accumulators).
__device__ __forceinline__ void wmma4(const __bf16* Alds, const __bf16* Blds,
                                      v8f acc[4], int lane, int wm, int wn) {
  const int r = lane & 15, hi = lane >> 4;
  v16bf a = frag_a(Alds + (wm * 16 + r) * LDT, hi);
#pragma unroll
  for (int s = 0; s < 4; ++s) {
    v16bf b = frag_b(Blds + (wn * 64 + s * 16 + r) * LDT, hi);
    acc[s] = __builtin_amdgcn_wmma_f32_16x16x32_bf16(false, a, false, b,
                                                     (short)0, acc[s], false, false);
  }
}

// ---- fp32 -> bf16 conversion ----------------------------------------------
__global__ __launch_bounds__(256) void k_f32_to_bf16(const float* __restrict__ s,
                                                     __bf16* __restrict__ d, long n) {
  long i = (long)blockIdx.x * 256 + threadIdx.x;
  if (i < n) d[i] = (__bf16)s[i];
}

// ---- Implicit-GEMM conv: out[n,co,h,w] = sum_{ci,kh,kw} w*x  (NCHW/OIHW) --
// Tile 64(M=Co) x 128(N=pixels), K in stages of 32, double-buffered LDS.
// grid.x = NPIX/128, grid.y = Co/64, 256 threads (8 waves).
__global__ __launch_bounds__(256)
void k_conv_wmma(const __bf16* __restrict__ X, const __bf16* __restrict__ Wt,
                 const float* __restrict__ bias, float* __restrict__ Out,
                 int Ci, int Co, int Kh, int pad, int dil) {
  __shared__ __bf16 Abuf[2][64 * LDT];
  __shared__ __bf16 Bbuf[2][128 * LDT];
  const int tid = threadIdx.x, lane = tid & 31, wave = tid >> 5;
  const int wm = wave & 3, wn = wave >> 2;
  const int nBase = blockIdx.x * 128, mBase = blockIdx.y * 64;
  const int KK = Kh * Kh;
  const int Ktot = Ci * KK;
  const int S = Ktot >> 5;
  // per-thread B-gather pixel (128 pixels, 2 threads per pixel)
  const int nl = tid & 127;
  const int p  = nBase + nl;
  const int n  = p >> 14;
  const int hw = p & 16383;
  const int h  = hw >> 7, w = hw & 127;
  const int kkb = tid >> 7;  // 0 or 1
  // per-thread A slot (64 rows x 32 K, one 16B chunk per thread)
  const int arow = tid >> 2, aseg = (tid & 3) * 8;
  v8f acc[4] = {{}, {}, {}, {}};

  auto loadA = [&](int st) {
    async_copy16(Wt + (long)(mBase + arow) * Ktot + (st << 5) + aseg,
                 Abuf[st & 1] + arow * LDT + aseg);
  };
  auto loadB = [&](int st) {  // padded im2col gather, stored transposed [n][k]
    __bf16* Bd = Bbuf[st & 1];
    const int k0 = st << 5;
#pragma unroll
    for (int i = 0; i < 16; ++i) {
      const int kk = kkb + i * 2;
      const int kg = k0 + kk;
      const int ci  = kg / KK;
      const int rem = kg - ci * KK;
      const int kh = rem / Kh, kw = rem - kh * Kh;
      const int ih = h - pad + kh * dil;
      const int iw = w - pad + kw * dil;
      __bf16 v = (__bf16)0.0f;
      if ((unsigned)ih < 128u && (unsigned)iw < 128u)
        v = X[((long)(n * Ci + ci) << 14) + (ih << 7) + iw];
      Bd[nl * LDT + kk] = v;
    }
  };

  loadA(0); loadB(0);
  for (int st = 0; st < S; ++st) {
    async_wait0();
    __syncthreads();
    if (st + 1 < S) { loadA(st + 1); loadB(st + 1); }
    wmma4(Abuf[st & 1], Bbuf[st & 1], acc, lane, wm, wn);
  }

  const int r = lane & 15, hi = lane >> 4;
#pragma unroll
  for (int e = 0; e < 8; ++e) {
    const int row = mBase + wm * 16 + e + hi * 8;  // C/D layout: M = e + 8*(lane>>4)
    const float bz = bias ? bias[row] : 0.0f;
#pragma unroll
    for (int s = 0; s < 4; ++s) {
      const int pc = nBase + wn * 64 + s * 16 + r;
      const int on = pc >> 14, ohw = pc & 16383;
      Out[((long)(on * Co + row) << 14) + ohw] = acc[s][e] + bz;
    }
  }
}

// ---- GroupNorm(16 groups) + ReLU, in place. grid = B*16 -------------------
__global__ __launch_bounds__(256)
void k_groupnorm_relu(float* __restrict__ buf, const float* __restrict__ scale,
                      const float* __restrict__ bias) {
  const int b = blockIdx.x >> 4, g = blockIdx.x & 15;
  float* base = buf + ((long)(b * 256 + g * 16) << 14);
  const int GS = 16 << 14;  // 262144 elems per group
  float s = 0.f, ss = 0.f;
  for (int i = threadIdx.x; i < GS; i += 256) { float v = base[i]; s += v; ss += v * v; }
  __shared__ float r0[256], r1[256];
  r0[threadIdx.x] = s; r1[threadIdx.x] = ss;
  __syncthreads();
  for (int st = 128; st > 0; st >>= 1) {
    if (threadIdx.x < st) { r0[threadIdx.x] += r0[threadIdx.x + st]; r1[threadIdx.x] += r1[threadIdx.x + st]; }
    __syncthreads();
  }
  const float mu  = r0[0] * (1.0f / GS);
  const float var = r1[0] * (1.0f / GS) - mu * mu;
  const float inv = rsqrtf(var + 1e-5f);
  for (int i = threadIdx.x; i < GS; i += 256) {
    const int c = g * 16 + (i >> 14);
    float v = (base[i] - mu) * inv * scale[c] + bias[c];
    base[i] = v > 0.f ? v : 0.f;
  }
}

// ---- L2 normalize along channel dim at each pixel, in place ---------------
__global__ __launch_bounds__(256)
void k_l2norm_ch(float* __restrict__ buf, int Cc) {
  const long idx = (long)blockIdx.x * 256 + threadIdx.x;  // (b,hw), 65536 total
  const int b = (int)(idx >> 14), hw = (int)(idx & 16383);
  float* base = buf + ((long)b * Cc << 14) + hw;
  float ss = 0.f;
  for (int c = 0; c < Cc; ++c) { float v = base[(long)c << 14]; ss += v * v; }
  const float inv = 1.0f / (sqrtf(ss) + 1e-10f);
  for (int c = 0; c < Cc; ++c) base[(long)c << 14] *= inv;
}

// ---- CC energy: att[b,h,w, 0:128]=eH(g), [128:256]=eW(v). grid=B*HW -------
__global__ __launch_bounds__(256)
void k_cc_energy(const float* __restrict__ qn, const float* __restrict__ kn,
                 float* __restrict__ att) {
  const int pix = blockIdx.x;
  const int b = pix >> 14, hw = pix & 16383;
  const int h = hw >> 7, w = hw & 127;
  __shared__ float qs[64];
  if (threadIdx.x < 64) qs[threadIdx.x] = qn[((long)(b * 64 + threadIdx.x) << 14) + hw];
  __syncthreads();
  const int g = threadIdx.x;
  const int off = (g < 128) ? ((g << 7) + w) : ((h << 7) + (g - 128));
  float e = 0.f;
  for (int c = 0; c < 64; ++c)
    e += qs[c] * kn[((long)(b * 64 + c) << 14) + off];
  att[((long)pix << 8) + g] = e;
}

// ---- softmax over last dim of length 256, one row per block ---------------
__global__ __launch_bounds__(256)
void k_softmax256(float* __restrict__ buf) {
  float* row = buf + ((long)blockIdx.x << 8);
  __shared__ float red[256];
  const float v = row[threadIdx.x];
  red[threadIdx.x] = v;
  __syncthreads();
  for (int st = 128; st > 0; st >>= 1) {
    if (threadIdx.x < st) red[threadIdx.x] = fmaxf(red[threadIdx.x], red[threadIdx.x + st]);
    __syncthreads();
  }
  const float m = red[0];
  __syncthreads();
  const float e = __expf(v - m);
  red[threadIdx.x] = e;
  __syncthreads();
  for (int st = 128; st > 0; st >>= 1) {
    if (threadIdx.x < st) red[threadIdx.x] += red[threadIdx.x + st];
    __syncthreads();
  }
  row[threadIdx.x] = e / red[0];
}

// ---- CC output: dst = outH + outW. grid=B*HW, thread = channel ------------
__global__ __launch_bounds__(256)
void k_cc_out(const float* __restrict__ v, const float* __restrict__ att,
              float* __restrict__ dst) {
  const int pix = blockIdx.x;
  const int b = pix >> 14, hw = pix & 16383;
  const int h = hw >> 7, w = hw & 127;
  __shared__ float sAtt[256];
  sAtt[threadIdx.x] = att[((long)pix << 8) + threadIdx.x];
  __syncthreads();
  const int c = threadIdx.x;
  const float* vb = v + ((long)(b * 256 + c) << 14);
  float o = 0.f;
  for (int g = 0; g < 128; ++g)  o += vb[(g << 7) + w] * sAtt[g];
  for (int x = 0; x < 128; ++x)  o += vb[(h << 7) + x] * sAtt[128 + x];
  dst[((long)(b * 256 + c) << 14) + hw] = o;
}

// ---- CAM: row-normalize f (over HW) and emit bf16. grid=B*C ---------------
__global__ __launch_bounds__(256)
void k_rownorm_bf16(const float* __restrict__ f, __bf16* __restrict__ fn) {
  const float* row = f + ((long)blockIdx.x << 14);
  __bf16* orow = fn + ((long)blockIdx.x << 14);
  __shared__ float red[256];
  float ss = 0.f;
  for (int i = threadIdx.x; i < 16384; i += 256) { float v = row[i]; ss += v * v; }
  red[threadIdx.x] = ss;
  __syncthreads();
  for (int st = 128; st > 0; st >>= 1) {
    if (threadIdx.x < st) red[threadIdx.x] += red[threadIdx.x + st];
    __syncthreads();
  }
  const float inv = 1.0f / (sqrtf(red[0]) + 1e-10f);
  for (int i = threadIdx.x; i < 16384; i += 256) orow[i] = (__bf16)(row[i] * inv);
}

// ---- energy[b] = fn[b](256xHW) . fn[b]^T : WMMA GEMM-NT, async double-buf -
// grid = (N/128=2, M/64=4, B); K = 16384 in 512 stages.
__global__ __launch_bounds__(256)
void k_gemm_nt_bf16(const __bf16* __restrict__ Abase, float* __restrict__ Ebase) {
  __shared__ __bf16 Abuf[2][64 * LDT];
  __shared__ __bf16 Bbuf[2][128 * LDT];
  const int b = blockIdx.z;
  const __bf16* A = Abase + ((long)b << 22);  // 256*16384
  const int mBase = blockIdx.y * 64, nBase = blockIdx.x * 128;
  const int tid = threadIdx.x, lane = tid & 31, wave = tid >> 5;
  const int wm = wave & 3, wn = wave >> 2;
  const int arow = tid >> 2, aseg = (tid & 3) * 8;
  const int brow = tid >> 1, bseg = (tid & 1) * 16;
  v8f acc[4] = {{}, {}, {}, {}};
  auto loadS = [&](int st) {
    const int k0 = st << 5;
    async_copy16(A + ((long)(mBase + arow) << 14) + k0 + aseg,
                 Abuf[st & 1] + arow * LDT + aseg);
    async_copy16(A + ((long)(nBase + brow) << 14) + k0 + bseg,
                 Bbuf[st & 1] + brow * LDT + bseg);
    async_copy16(A + ((long)(nBase + brow) << 14) + k0 + bseg + 8,
                 Bbuf[st & 1] + brow * LDT + bseg + 8);
  };
  loadS(0);
  for (int st = 0; st < 512; ++st) {
    async_wait0();
    __syncthreads();
    if (st + 1 < 512) loadS(st + 1);
    wmma4(Abuf[st & 1], Bbuf[st & 1], acc, lane, wm, wn);
  }
  float* E = Ebase + (long)b * 65536;
  const int r = lane & 15, hi = lane >> 4;
#pragma unroll
  for (int e = 0; e < 8; ++e) {
    const int row = mBase + wm * 16 + e + hi * 8;
#pragma unroll
    for (int s = 0; s < 4; ++s)
      E[row * 256 + nBase + wn * 64 + s * 16 + r] = acc[s][e];
  }
}

// ---- out = att[b](256x256) . f[b](256xHW); fused: d_out = gamma*out + x ---
// grid = (HW/128=128, C/64=4, B); K = 256 in 8 stages.
__global__ __launch_bounds__(256)
void k_gemm_nn_cam(const __bf16* __restrict__ attb, const __bf16* __restrict__ fb,
                   const float* __restrict__ xin, const float* __restrict__ gm,
                   int slot, float* __restrict__ out) {
  __shared__ __bf16 Abuf[2][64 * LDT];
  __shared__ __bf16 Bbuf[2][128 * LDT];
  const int b = blockIdx.z;
  const __bf16* A  = attb + (long)b * 65536;  // 256x256 row-major
  const __bf16* Bm = fb + ((long)b << 22);    // 256xHW row-major
  const int mBase = blockIdx.y * 64, nBase = blockIdx.x * 128;
  const int tid = threadIdx.x, lane = tid & 31, wave = tid >> 5;
  const int wm = wave & 3, wn = wave >> 2;
  const int arow = tid >> 2, aseg = (tid & 3) * 8;
  const int nl = tid & 127, kkb = tid >> 7;
  v8f acc[4] = {{}, {}, {}, {}};
  auto loadA = [&](int st) {
    async_copy16(A + (mBase + arow) * 256 + (st << 5) + aseg,
                 Abuf[st & 1] + arow * LDT + aseg);
  };
  auto loadB = [&](int st) {  // rows of f, stored transposed [n][k]
    __bf16* Bd = Bbuf[st & 1];
#pragma unroll
    for (int i = 0; i < 16; ++i) {
      const int kk = kkb + i * 2;
      Bd[nl * LDT + kk] = Bm[((long)((st << 5) + kk) << 14) + nBase + nl];
    }
  };
  loadA(0); loadB(0);
  for (int st = 0; st < 8; ++st) {
    async_wait0();
    __syncthreads();
    if (st + 1 < 8) { loadA(st + 1); loadB(st + 1); }
    wmma4(Abuf[st & 1], Bbuf[st & 1], acc, lane, wm, wn);
  }
  const float gamma = gm[slot];
  const int r = lane & 15, hi = lane >> 4;
#pragma unroll
  for (int e = 0; e < 8; ++e) {
    const int row = mBase + wm * 16 + e + hi * 8;
#pragma unroll
    for (int s = 0; s < 4; ++s) {
      const int col = nBase + wn * 64 + s * 16 + r;
      const long xi = ((long)(b * 256 + row) << 14) + col;
      out[((long)(slot * 1024 + b * 256 + row) << 14) + col] = gamma * acc[s][e] + xin[xi];
    }
  }
}

// ---------------------------------------------------------------------------
extern "C" void kernel_launch(void* const* d_in, const int* in_sizes, int n_in,
                              void* d_out, int out_size, void* d_ws, size_t ws_size,
                              hipStream_t stream) {
  (void)in_sizes; (void)n_in; (void)out_size; (void)ws_size;
  const long NEL = 16777216;   // B*C*HW = 4*256*16384
  const int  NPIX = 65536;     // B*HW

  // inputs per setup_inputs() order
  const float* x  = (const float*)d_in[0];
  const float* wS[5] = {(const float*)d_in[1], (const float*)d_in[2], (const float*)d_in[3],
                        (const float*)d_in[4], (const float*)d_in[5]};
  const float* sS[5] = {(const float*)d_in[6], (const float*)d_in[7], (const float*)d_in[8],
                        (const float*)d_in[9], (const float*)d_in[10]};
  const float* bS[5] = {(const float*)d_in[11], (const float*)d_in[12], (const float*)d_in[13],
                        (const float*)d_in[14], (const float*)d_in[15]};
  const float* wq = (const float*)d_in[16]; const float* bq = (const float*)d_in[17];
  const float* wk = (const float*)d_in[18]; const float* bk = (const float*)d_in[19];
  const float* wv = (const float*)d_in[20]; const float* bv = (const float*)d_in[21];
  const float* gammas = (const float*)d_in[22];

  // workspace carve
  char* ws = (char*)d_ws; size_t off = 0;
  auto carve = [&](size_t bytes) -> void* {
    off = (off + 255) & ~(size_t)255; void* p = ws + off; off += bytes; return p;
  };
  __bf16* bfA = (__bf16*)carve(NEL * 2);          // bf16 scratch A (x, cc input, fn)
  __bf16* bfB = (__bf16*)carve(NEL * 2);          // bf16 scratch B (f for CAM)
  __bf16* wbf = (__bf16*)carve(2523136UL * 2);    // all weights bf16
  float* br[5];
  for (int i = 0; i < 5; ++i) br[i] = (float*)carve(NEL * 4);
  float* qb   = (float*)carve((long)4 * 64 * 16384 * 4);
  float* kb   = (float*)carve((long)4 * 64 * 16384 * 4);
  float* vb   = (float*)carve(NEL * 4);
  float* attf = (float*)carve(NEL * 4);           // B*HW*256
  float* tmpf = (float*)carve(NEL * 4);
  float* energ = (float*)carve((long)4 * 256 * 256 * 4);
  __bf16* attbf = (__bf16*)carve((long)4 * 256 * 256 * 2);

  // bf16 weight segment offsets (elements; all multiples of 8 -> 16B aligned)
  const long ow[5] = {0, 65536, 655360, 1245184, 1835008};  // w1, w2, w3, w4, wc
  const long owq = 2424832, owk = 2441216, owv = 2457600;
  const long wcnt[5] = {65536, 589824, 589824, 589824, 589824};

  dim3 blk(256);
  // --- convert x and weights to bf16
  k_f32_to_bf16<<<(unsigned)(NEL / 256), blk, 0, stream>>>(x, bfA, NEL);
  for (int i = 0; i < 5; ++i)
    k_f32_to_bf16<<<(unsigned)((wcnt[i] + 255) / 256), blk, 0, stream>>>(wS[i], wbf + ow[i], wcnt[i]);
  k_f32_to_bf16<<<64, blk, 0, stream>>>(wq, wbf + owq, 16384);
  k_f32_to_bf16<<<64, blk, 0, stream>>>(wk, wbf + owk, 16384);
  k_f32_to_bf16<<<256, blk, 0, stream>>>(wv, wbf + owv, 65536);

  // --- ASPP branches: conv (implicit-GEMM WMMA) + GroupNorm + ReLU
  const int pads[5] = {0, 6, 12, 18, 1};
  const int dils[5] = {1, 6, 12, 18, 1};
  const int kszs[5] = {1, 3, 3, 3, 3};
  for (int i = 0; i < 5; ++i) {
    k_conv_wmma<<<dim3(NPIX / 128, 4), blk, 0, stream>>>(bfA, wbf + ow[i], nullptr, br[i],
                                                         256, 256, kszs[i], pads[i], dils[i]);
    k_groupnorm_relu<<<64, blk, 0, stream>>>(br[i], sS[i], bS[i]);
  }

  // --- criss-cross attention x2 on branch 5 (ping-pong br[4] <-> tmpf)
  float* cur = br[4];
  float* nxt = tmpf;
  for (int it = 0; it < 2; ++it) {
    k_f32_to_bf16<<<(unsigned)(NEL / 256), blk, 0, stream>>>(cur, bfA, NEL);
    k_conv_wmma<<<dim3(NPIX / 128, 1), blk, 0, stream>>>(bfA, wbf + owq, bq, qb, 256, 64, 1, 0, 1);
    k_conv_wmma<<<dim3(NPIX / 128, 1), blk, 0, stream>>>(bfA, wbf + owk, bk, kb, 256, 64, 1, 0, 1);
    k_conv_wmma<<<dim3(NPIX / 128, 4), blk, 0, stream>>>(bfA, wbf + owv, bv, vb, 256, 256, 1, 0, 1);
    k_l2norm_ch<<<NPIX / 256, blk, 0, stream>>>(qb, 64);
    k_l2norm_ch<<<NPIX / 256, blk, 0, stream>>>(kb, 64);
    k_cc_energy<<<NPIX, blk, 0, stream>>>(qb, kb, attf);
    k_softmax256<<<NPIX, blk, 0, stream>>>(attf);
    k_cc_out<<<NPIX, blk, 0, stream>>>(vb, attf, nxt);
    float* t = cur; cur = nxt; nxt = t;   // after 2 iters result is in br[4]
  }

  // --- CAM per branch: energy = fn.fnT (WMMA NT) -> softmax -> att@f (WMMA NN,
  //     fused gamma*out + x epilogue into d_out)
  for (int i = 0; i < 5; ++i) {
    k_rownorm_bf16<<<1024, blk, 0, stream>>>(br[i], bfA);
    k_gemm_nt_bf16<<<dim3(2, 4, 4), blk, 0, stream>>>(bfA, energ);
    k_softmax256<<<1024, blk, 0, stream>>>(energ);
    k_f32_to_bf16<<<1024, blk, 0, stream>>>(energ, attbf, 262144);
    k_f32_to_bf16<<<(unsigned)(NEL / 256), blk, 0, stream>>>(br[i], bfB, NEL);
    k_gemm_nn_cam<<<dim3(128, 4, 4), blk, 0, stream>>>(attbf, bfB, br[i], gammas, i,
                                                       (float*)d_out);
  }
}
